// QuantizedLinear_29222957482640
// MI455X (gfx1250) — 8-bit matrix operand order repaired and checked by reading, compiled, not run on hardware
//
#include <hip/hip_runtime.h>

typedef __attribute__((ext_vector_type(8))) int   v8i;
typedef __attribute__((ext_vector_type(4))) int   v4i;
typedef __attribute__((ext_vector_type(2))) int   v2i;
typedef __attribute__((ext_vector_type(4))) float v4f;

#define N_ROWS  4096
#define IN_DIM  4096
#define OUT_DIM 4096

#define BM 256
#define BN 128
#define BK 64

__global__ __launch_bounds__(256) void quant_act_kernel(
    const float* __restrict__ x, const float* __restrict__ scale_p,
    const int* __restrict__ zp_p, signed char* __restrict__ xq) {
  const int idx = (blockIdx.x * 256 + threadIdx.x) * 16;
  const float inv = 1.0f / scale_p[0];
  const int   zp  = zp_p[0];
  const v4f* xv = (const v4f*)(x + idx);
  v4i packed;
#pragma unroll
  for (int j = 0; j < 4; ++j) {
    v4f v = __builtin_nontemporal_load(xv + j);
    int b[4];
#pragma unroll
    for (int e = 0; e < 4; ++e) {
      int q = (int)rintf(v[e] * inv) + zp;
      q = q < 0 ? 0 : (q > 255 ? 255 : q);
      b[e] = (q - zp) & 0xff;
    }
    packed[j] = b[0] | (b[1] << 8) | (b[2] << 16) | (b[3] << 24);
  }
  *(v4i*)(xq + idx) = packed;
}

__global__ __launch_bounds__(256) void quant_wgt_kernel(
    const float* __restrict__ w, const float* __restrict__ wscale,
    const int* __restrict__ wzp, signed char* __restrict__ wq) {
  const int idx = (blockIdx.x * 256 + threadIdx.x) * 16;
  const int row = idx >> 12;
  const float inv = 1.0f / wscale[row];
  const int   zp  = wzp[row];
  const v4f* wv = (const v4f*)(w + idx);
  v4i packed;
#pragma unroll
  for (int j = 0; j < 4; ++j) {
    v4f v = __builtin_nontemporal_load(wv + j);
    int b[4];
#pragma unroll
    for (int e = 0; e < 4; ++e) {
      int q = (int)rintf(v[e] * inv) + zp;
      q = q < 0 ? 0 : (q > 255 ? 255 : q);
      b[e] = (q - zp) & 0xff;
    }
    packed[j] = b[0] | (b[1] << 8) | (b[2] << 16) | (b[3] << 24);
  }
  *(v4i*)(wq + idx) = packed;
}

__global__ __launch_bounds__(256) void gemm_i8_wmma_kernel(
    const signed char* __restrict__ xq, const signed char* __restrict__ wq,
    const float* __restrict__ bias, const float* __restrict__ act_scale_p,
    const float* __restrict__ wscale, float* __restrict__ out) {
  const int lane = threadIdx.x & 31;
  const int wave = threadIdx.x >> 5;
  const int l16  = lane & 15;
  const int hi   = lane >> 4;
  const int waveM = wave & 3;
  const int waveN = wave >> 2;
  const int rowBase = blockIdx.y * BM + waveM * 64;
  const int colBase = blockIdx.x * BN + waveN * 64;

  v8i acc[4][4];
#pragma unroll
  for (int mt = 0; mt < 4; ++mt)
#pragma unroll
    for (int nt = 0; nt < 4; ++nt)
#pragma unroll
      for (int r = 0; r < 8; ++r) acc[mt][nt][r] = 0;

  for (int k0 = 0; k0 < IN_DIM; k0 += BK) {
    v8i afrag[4];
#pragma unroll
    for (int mt = 0; mt < 4; ++mt) {
      const signed char* ap =
          xq + (size_t)(rowBase + mt * 16 + l16) * IN_DIM + k0 + hi * 16;
      v2i c0 = *(const v2i*)(ap + 0);
      v2i c1 = *(const v2i*)(ap + 8);
      v2i c2 = *(const v2i*)(ap + 32);
      v2i c3 = *(const v2i*)(ap + 40);
      v8i a;
      a[0] = c0.x; a[1] = c0.y;
      a[2] = c1.x; a[3] = c1.y;
      a[4] = c2.x; a[5] = c2.y;
      a[6] = c3.x; a[7] = c3.y;
      afrag[mt] = a;
    }
#pragma unroll
    for (int nt = 0; nt < 4; ++nt) {
      const signed char* bp =
          wq + (size_t)(colBase + nt * 16 + l16) * IN_DIM + k0 + hi * 16;
      v4i b0 = *(const v4i*)(bp + 0);
      v4i b1 = *(const v4i*)(bp + 32);
      v8i b;
      b[0] = b0.x; b[1] = b0.y; b[2] = b0.z; b[3] = b0.w;
      b[4] = b1.x; b[5] = b1.y; b[6] = b1.z; b[7] = b1.w;
#pragma unroll
      for (int mt = 0; mt < 4; ++mt) {
        acc[mt][nt] = __builtin_amdgcn_wmma_i32_16x16x64_iu8(
            true, afrag[mt], true, b, acc[mt][nt], false, false);
      }
    }
  }

  const float s_act = act_scale_p[0];
#pragma unroll
  for (int nt = 0; nt < 4; ++nt) {
    const int o  = colBase + nt * 16 + l16;
    const float sc = s_act * wscale[o];
    const float bo = bias[o];
#pragma unroll
    for (int mt = 0; mt < 4; ++mt) {
#pragma unroll
      for (int r = 0; r < 8; ++r) {
        const int row = rowBase + mt * 16 + hi * 8 + r;
        __builtin_nontemporal_store((float)acc[mt][nt][r] * sc + bo,
                                    &out[(size_t)row * OUT_DIM + o]);
      }
    }
  }
}

extern "C" void kernel_launch(void* const* d_in, const int* in_sizes, int n_in,
                              void* d_out, int out_size, void* d_ws, size_t ws_size,
                              hipStream_t stream) {
  const float* x        = (const float*)d_in[0];
  const float* weight   = (const float*)d_in[1];
  const float* bias     = (const float*)d_in[2];
  const float* act_s    = (const float*)d_in[3];
  const int*   act_zp   = (const int*)d_in[4];
  const float* w_s      = (const float*)d_in[5];
  const int*   w_zp     = (const int*)d_in[6];
  float*       out      = (float*)d_out;

  signed char* xq = (signed char*)d_ws;
  signed char* wq = (signed char*)d_ws + (size_t)N_ROWS * IN_DIM;

  const int total = N_ROWS * IN_DIM;
  const int qblocks = total / (16 * 256);

  quant_act_kernel<<<qblocks, 256, 0, stream>>>(x, act_s, act_zp, xq);
  quant_wgt_kernel<<<qblocks, 256, 0, stream>>>(weight, w_s, w_zp, wq);

  dim3 grid(OUT_DIM / BN, N_ROWS / BM);
  gemm_i8_wmma_kernel<<<grid, 256, 0, stream>>>(xq, wq, bias, act_s, w_s, out);
}
